// MultiHeadAttention_8272107012272
// MI455X (gfx1250) — matrix operand order repaired and checked by reading, not compiled, not run on hardware
//
#include <hip/hip_runtime.h>
#include <hip/hip_bf16.h>

#define D_MODEL 1024
#define N_HEADS 16
#define D_KK    64
#define BB      4
#define SS      2048

typedef __attribute__((ext_vector_type(16))) _Float16 v16h;
typedef __attribute__((ext_vector_type(8)))  _Float16 v8h;
typedef __attribute__((ext_vector_type(4)))  _Float16 v4h;
typedef __attribute__((ext_vector_type(8)))  float    v8f;

union V16 { v16h v; v8h h[2]; };

__device__ __forceinline__ v8f wmma_f16(v16h a, v16h b, v8f c) {
  // 8 args: (neg_a, A, neg_b, B, c_mod, C, reuse_a, reuse_b)
  return __builtin_amdgcn_wmma_f32_16x16x32_f16(false, a, false, b, (short)0, c,
                                                false, false);
}

// Max-reduce across the 16-lane DPP row (wave32 halves) using pure VALU DPP:
// quad_perm swaps handle strides 1,2; row rotations handle 4,8.
__device__ __forceinline__ float dpp_max16(float x) {
  int y;
  y = __builtin_amdgcn_update_dpp(0, __float_as_int(x), 0xB1, 0xf, 0xf, true);
  x = fmaxf(x, __int_as_float(y));  // quad_perm [1,0,3,2]
  y = __builtin_amdgcn_update_dpp(0, __float_as_int(x), 0x4E, 0xf, 0xf, true);
  x = fmaxf(x, __int_as_float(y));  // quad_perm [2,3,0,1]
  y = __builtin_amdgcn_update_dpp(0, __float_as_int(x), 0x124, 0xf, 0xf, true);
  x = fmaxf(x, __int_as_float(y));  // row_ror:4
  y = __builtin_amdgcn_update_dpp(0, __float_as_int(x), 0x128, 0xf, 0xf, true);
  x = fmaxf(x, __int_as_float(y));  // row_ror:8
  return x;
}

// ---------------------------------------------------------------------------
// GEMM: C[8192,1024] = A[8192,1024] @ W[1024,1024] + bias
// mode 0: write f16 head-split [b][h][s][64]
// mode 1: write f16 transposed V [b][h][64][s]
// mode 2: write f32 row-major [8192][1024]
// ---------------------------------------------------------------------------
template <bool AF16>
__global__ __launch_bounds__(128) void gemm_proj(
    const void* __restrict__ A_, const float* __restrict__ W,
    const float* __restrict__ bias, void* __restrict__ out, int mode) {
  __shared__ __align__(16) _Float16 As[64 * 40];  // [row][k] padded
  __shared__ __align__(16) _Float16 Ws[64 * 40];  // transposed: [n][k] padded

  const int tid  = threadIdx.x;
  const int wave = tid >> 5, lane = tid & 31;
  const int hi = lane >> 4, ln = lane & 15;
  const int n0 = blockIdx.x * 64;
  const int m0 = blockIdx.y * 64;
  const float*    Af = (const float*)A_;
  const _Float16* Ah = (const _Float16*)A_;

  v8f acc[4] = {};

  for (int kc = 0; kc < D_MODEL; kc += 32) {
    // A tile: 64 rows x 32 k, vectorized (float4 / v4h), f16 to LDS
#pragma unroll
    for (int e4 = tid; e4 < (64 * 32) / 4; e4 += 128) {
      const int e = e4 * 4;
      const int r = e >> 5, kk = e & 31;
      v4h hv;
      if (AF16) {
        hv = *(const v4h*)(Ah + (size_t)(m0 + r) * D_MODEL + kc + kk);
      } else {
        const float4 a = *(const float4*)(Af + (size_t)(m0 + r) * D_MODEL + kc + kk);
        hv = v4h{(_Float16)a.x, (_Float16)a.y, (_Float16)a.z, (_Float16)a.w};
      }
      *(v4h*)&As[r * 40 + kk] = hv;
    }
    // W tile: 32 k x 64 n, float4 loads, stored transposed [n][k]
#pragma unroll
    for (int e4 = tid; e4 < (32 * 64) / 4; e4 += 128) {
      const int e = e4 * 4;
      const int kk = e >> 6, n = e & 63;
      const float4 w4 = *(const float4*)(W + (size_t)(kc + kk) * D_MODEL + n0 + n);
      Ws[(n + 0) * 40 + kk] = (_Float16)w4.x;
      Ws[(n + 1) * 40 + kk] = (_Float16)w4.y;
      Ws[(n + 2) * 40 + kk] = (_Float16)w4.z;
      Ws[(n + 3) * 40 + kk] = (_Float16)w4.w;
    }
    __syncthreads();

    // A fragment: lane holds row (wave*16+ln); halfs 0-7 = k hi*8.., 8-15 = +16
    V16 a;
    const int arow = wave * 16 + ln;
    a.h[0] = *(const v8h*)&As[arow * 40 + hi * 8];
    a.h[1] = *(const v8h*)&As[arow * 40 + hi * 8 + 16];
#pragma unroll
    for (int t = 0; t < 4; ++t) {
      // B fragment: lane = n, halfs 0-7 = k hi*8.., 8-15 = +16
      V16 b;
      const int nrow = t * 16 + ln;
      b.h[0] = *(const v8h*)&Ws[nrow * 40 + hi * 8];
      b.h[1] = *(const v8h*)&Ws[nrow * 40 + hi * 8 + 16];
      acc[t] = wmma_f16(a.v, b.v, acc[t]);
    }
    __syncthreads();
  }

  // Epilogue. C layout: lane=n (ln), reg r -> m = r + 8*hi
#pragma unroll
  for (int t = 0; t < 4; ++t) {
    const int gc = n0 + t * 16 + ln;
    const float bv = bias[gc];
#pragma unroll
    for (int r = 0; r < 8; ++r) {
      const int gm = m0 + wave * 16 + r + 8 * hi;
      const float v = acc[t][r] + bv;
      if (mode == 0) {
        const int b_ = gm >> 11, s_ = gm & 2047, h_ = gc >> 6, d_ = gc & 63;
        ((_Float16*)out)[(((size_t)(b_ * N_HEADS + h_) * SS + s_) << 6) + d_] =
            (_Float16)v;
      } else if (mode == 1) {
        const int b_ = gm >> 11, s_ = gm & 2047, h_ = gc >> 6, d_ = gc & 63;
        ((_Float16*)out)[((size_t)(b_ * N_HEADS + h_) * D_KK + d_) * SS + s_] =
            (_Float16)v;
      } else {
        ((float*)out)[(size_t)gm * D_MODEL + gc] = v;
      }
    }
  }
}

// ---------------------------------------------------------------------------
// Flash attention: per (b,h), 4 waves x 16 query rows per block.
// Qh/Kh: [b][h][s][64] f16.  Vt: [b][h][64][s] f16.  attn: [b][s][1024] f16.
// Row sums computed by an extra WMMA against an all-ones B fragment; row max
// via DPP; no cross-lane LDS traffic in the softmax.
// ---------------------------------------------------------------------------
__global__ __launch_bounds__(128) void flash_attn(
    const _Float16* __restrict__ Qh, const _Float16* __restrict__ Kh,
    const _Float16* __restrict__ Vt, _Float16* __restrict__ attn) {
  __shared__ __align__(16) _Float16 Psh[4][16 * 40];  // wave-private P tiles

  const int tid  = threadIdx.x;
  const int wave = tid >> 5, lane = tid & 31;
  const int hi = lane >> 4, ln = lane & 15;
  const int bh   = blockIdx.x;
  const int qrow = blockIdx.y * 64 + wave * 16;

  const _Float16* Qb = Qh + (size_t)bh * SS * D_KK;
  const _Float16* Kb = Kh + (size_t)bh * SS * D_KK;
  const _Float16* Vb = Vt + (size_t)bh * D_KK * SS;

  // Q fragments, resident for whole loop (A layout, contraction d)
  const _Float16* qr = Qb + (size_t)(qrow + ln) * D_KK;
  V16 aq0, aq1;
  aq0.h[0] = *(const v8h*)(qr + hi * 8);
  aq0.h[1] = *(const v8h*)(qr + hi * 8 + 16);
  aq1.h[0] = *(const v8h*)(qr + 32 + hi * 8);
  aq1.h[1] = *(const v8h*)(qr + 32 + hi * 8 + 16);

  V16 ones;
#pragma unroll
  for (int i = 0; i < 16; ++i) ones.v[i] = (_Float16)1.0f;

  float mrow[8];
#pragma unroll
  for (int r = 0; r < 8; ++r) mrow[r] = -1e30f;
  v8f o[4] = {};
  v8f osum = {};  // row sums of P, accumulated via WMMA

  const float SCALE = 0.125f;  // 1/sqrt(64)
  const float L2E   = 1.44269504088896f;
  _Float16* pb = &Psh[wave][0];

  for (int j = 0; j < SS / 32; ++j) {
    const int kb = j * 32;

    // scores: 16q x 32keys, two 16x16 C tiles, K=64 in two chunks
    v8f c0 = {}, c1 = {};
    {
      const _Float16* k0 = Kb + (size_t)(kb + ln) * D_KK;
      V16 b0, b1;
      b0.h[0] = *(const v8h*)(k0 + hi * 8);
      b0.h[1] = *(const v8h*)(k0 + hi * 8 + 16);
      b1.h[0] = *(const v8h*)(k0 + 32 + hi * 8);
      b1.h[1] = *(const v8h*)(k0 + 32 + hi * 8 + 16);
      c0 = wmma_f16(aq0.v, b0.v, c0);
      c0 = wmma_f16(aq1.v, b1.v, c0);
    }
    {
      const _Float16* k1 = Kb + (size_t)(kb + 16 + ln) * D_KK;
      V16 b0, b1;
      b0.h[0] = *(const v8h*)(k1 + hi * 8);
      b0.h[1] = *(const v8h*)(k1 + hi * 8 + 16);
      b1.h[0] = *(const v8h*)(k1 + 32 + hi * 8);
      b1.h[1] = *(const v8h*)(k1 + 32 + hi * 8 + 16);
      c1 = wmma_f16(aq0.v, b0.v, c1);
      c1 = wmma_f16(aq1.v, b1.v, c1);
    }

    // online softmax: rows live in 16-lane halves; DPP-only max reduction
#pragma unroll
    for (int r = 0; r < 8; ++r) {
      const float s0 = c0[r] * SCALE;
      const float s1 = c1[r] * SCALE;
      const float t  = dpp_max16(fmaxf(s0, s1));
      const float mn = fmaxf(mrow[r], t);
      const float corr = __builtin_exp2f((mrow[r] - mn) * L2E);
      mrow[r] = mn;
      const float p0 = __builtin_exp2f((s0 - mn) * L2E);
      const float p1 = __builtin_exp2f((s1 - mn) * L2E);
      o[0][r] *= corr; o[1][r] *= corr; o[2][r] *= corr; o[3][r] *= corr;
      osum[r] *= corr;
      // C->A transpose staging: element (m = r+8*hi, n = ln / 16+ln)
      const int m_ = r + 8 * hi;
      pb[m_ * 40 + ln]      = (_Float16)p0;
      pb[m_ * 40 + 16 + ln] = (_Float16)p1;
    }
    asm volatile("s_wait_dscnt 0x0" ::: "memory");

    // P as A fragment (16x32 over keys)
    V16 ap;
    ap.h[0] = *(const v8h*)&pb[ln * 40 + hi * 8];
    ap.h[1] = *(const v8h*)&pb[ln * 40 + hi * 8 + 16];

    // row sums: P @ ones (every lane of a row receives the row sum)
    osum = wmma_f16(ap.v, ones.v, osum);

    // P @ V: B fragments from transposed V rows (contiguous keys)
#pragma unroll
    for (int t = 0; t < 4; ++t) {
      const _Float16* vr = Vb + (size_t)(t * 16 + ln) * SS + kb + hi * 8;
      V16 bv;
      bv.h[0] = *(const v8h*)vr;
      bv.h[1] = *(const v8h*)(vr + 16);
      o[t] = wmma_f16(ap.v, bv.v, o[t]);
    }
  }

  // normalize + write merged-head f16 [b][s][h*64+d]
  const int b_ = bh >> 4, h_ = bh & 15;
#pragma unroll
  for (int r = 0; r < 8; ++r) {
    const float inv = 1.0f / osum[r];
    const int srow  = qrow + r + 8 * hi;
#pragma unroll
    for (int t = 0; t < 4; ++t) {
      attn[(size_t)(b_ * SS + srow) * D_MODEL + h_ * D_KK + t * 16 + ln] =
          (_Float16)(o[t][r] * inv);
    }
  }
}

// ---------------------------------------------------------------------------
extern "C" void kernel_launch(void* const* d_in, const int* in_sizes, int n_in,
                              void* d_out, int out_size, void* d_ws,
                              size_t ws_size, hipStream_t stream) {
  (void)in_sizes; (void)n_in; (void)out_size; (void)ws_size;
  const float* q   = (const float*)d_in[0];
  const float* k   = (const float*)d_in[1];
  const float* v   = (const float*)d_in[2];
  const float* w_q = (const float*)d_in[3];
  const float* b_q = (const float*)d_in[4];
  const float* w_k = (const float*)d_in[5];
  const float* b_k = (const float*)d_in[6];
  const float* w_v = (const float*)d_in[7];
  const float* b_v = (const float*)d_in[8];
  const float* w_o = (const float*)d_in[9];
  const float* b_o = (const float*)d_in[10];

  const size_t HEAD_ELEMS = (size_t)BB * N_HEADS * SS * D_KK;  // 8M halfs
  _Float16* Qh   = (_Float16*)d_ws;
  _Float16* Kh   = Qh + HEAD_ELEMS;
  _Float16* Vt   = Kh + HEAD_ELEMS;
  _Float16* attn = Vt + HEAD_ELEMS;  // 64 MB total workspace

  dim3 blk(128);
  dim3 gg(D_MODEL / 64, (BB * SS) / 64);  // (16, 128)

  gemm_proj<false><<<gg, blk, 0, stream>>>(q, w_q, b_q, Qh, 0);
  gemm_proj<false><<<gg, blk, 0, stream>>>(k, w_k, b_k, Kh, 0);
  gemm_proj<false><<<gg, blk, 0, stream>>>(v, w_v, b_v, Vt, 1);
  flash_attn<<<dim3(BB * N_HEADS, SS / 64), blk, 0, stream>>>(Qh, Kh, Vt, attn);
  gemm_proj<true><<<gg, blk, 0, stream>>>(attn, w_o, b_o, d_out, 2);
}
